// multihead_attention_28020366639529
// MI455X (gfx1250) — matrix operand order repaired and checked by reading, not compiled, not run on hardware
//
#include <hip/hip_runtime.h>

// ---------------------------------------------------------------------------
// Multi-head attention block for gfx1250 (MI455X), f16 WMMA everywhere.
//   DIM=1024, B=2, T=2048, H=16, Dh=64
// GEMMs: 64x256 block tile, 8 waves, async-DMA (ASYNCcnt) double-buffered LDS.
// Attention: flash-style, per-wave 16-query tile, scores never hit memory.
// ---------------------------------------------------------------------------

typedef __attribute__((ext_vector_type(16))) _Float16 v16h;
typedef __attribute__((ext_vector_type(8)))  _Float16 v8h;
typedef __attribute__((ext_vector_type(8)))  float    v8f;

#define DIMC   1024
#define TSEQ   2048
#define NBATCH 2
#define NHEAD  16
#define HDIM   64
#define BT     (NBATCH * TSEQ)      /* 4096 rows */

static __device__ inline v8f wmma_f16(v16h a, v16h b, v8f c) {
  return __builtin_amdgcn_wmma_f32_16x16x32_f16(false, a, false, b, (short)0, c,
                                                false, false);
}

// Build a 16x32 f16 A-fragment:
//   lanes 0-15  : row m, K = {0..15}   (akoff = 0)
//   lanes 16-31 : row m, K = {16..31}  (akoff = 8)
// `p` points at row m, column k0 of a row-major f16 matrix (global or LDS).
static __device__ inline v16h load_afrag(const _Float16* p, int akoff) {
  v8h lo = *(const v8h*)(p + 2 * akoff);
  v8h hh = *(const v8h*)(p + 2 * akoff + 8);
  v16h a;
#pragma unroll
  for (int i = 0; i < 8; ++i) { a[i] = lo[i]; a[8 + i] = hh[i]; }
  return a;
}

// CDNA5 async DMA: 16B global -> LDS, tracked by ASYNCcnt (not LOADcnt).
static __device__ inline void async_b128(unsigned lds_byte_off,
                                         const _Float16* g) {
  asm volatile("global_load_async_to_lds_b128 %0, %1, off"
               :: "v"(lds_byte_off), "v"(g)
               : "memory");
}
static __device__ inline void wait_async0() {
  asm volatile("s_wait_asynccnt 0" ::: "memory");
}

// ---------------------------------------------------------------------------
// fp32 -> f16 conversion of the four 1024x1024 weight matrices
// ---------------------------------------------------------------------------
__global__ __launch_bounds__(256) void cvt4_kernel(
    const float* __restrict__ a, const float* __restrict__ b,
    const float* __restrict__ c, const float* __restrict__ d,
    _Float16* __restrict__ oa, _Float16* __restrict__ ob,
    _Float16* __restrict__ oc, _Float16* __restrict__ od) {
  int i = blockIdx.x * 256 + threadIdx.x;    // grid covers 1024*1024
  oa[i] = (_Float16)a[i];
  ob[i] = (_Float16)b[i];
  oc[i] = (_Float16)c[i];
  od[i] = (_Float16)d[i];
}

// ---------------------------------------------------------------------------
// LayerNorm (fp32 in) -> f16 out, one block per row of 1024
// ---------------------------------------------------------------------------
__global__ __launch_bounds__(128) void ln_f16_kernel(
    const float* __restrict__ x, const float* __restrict__ g,
    const float* __restrict__ bb, _Float16* __restrict__ h16) {
  __shared__ float rs[128];
  __shared__ float rss[128];
  const int row = blockIdx.x;
  const float* xr = x + (size_t)row * DIMC;
  float s = 0.f, s2 = 0.f;
  for (int i = threadIdx.x; i < DIMC; i += 128) {
    float v = xr[i];
    s += v;
    s2 += v * v;
  }
  rs[threadIdx.x] = s;
  rss[threadIdx.x] = s2;
  __syncthreads();
  for (int off = 64; off > 0; off >>= 1) {
    if (threadIdx.x < off) {
      rs[threadIdx.x] += rs[threadIdx.x + off];
      rss[threadIdx.x] += rss[threadIdx.x + off];
    }
    __syncthreads();
  }
  const float mu = rs[0] * (1.0f / DIMC);
  const float var = rss[0] * (1.0f / DIMC) - mu * mu;
  const float rstd = rsqrtf(var + 1e-5f);
  for (int i = threadIdx.x; i < DIMC; i += 128) {
    h16[(size_t)row * DIMC + i] =
        (_Float16)((xr[i] - mu) * rstd * g[i] + bb[i]);
  }
}

// ---------------------------------------------------------------------------
// WMMA GEMM: Y[m][n] = sum_k A[m][k] * W[n][k] + bias[n]   (torch Linear)
//
// Block = 256 threads (8 waves), block tile 64(M) x 256(N).
// Per k-chunk (32): A 64x32 (4KB) + B 256x32 (16KB) staged into LDS via
// global_load_async_to_lds_b128, double-buffered on ASYNCcnt.
// Wave (wm,wn) computes 32x64 = 2x4 grid of 16x16 WMMA tiles.
//
// MODE 0: f16 out, head-major layout [bh][t][d]            (Q, K)
// MODE 1: f16 out, transposed head layout [bh][d][t]       (V^T)
// MODE 2: fp32 out, row-major [m][n]                       (final proj)
// ---------------------------------------------------------------------------
template <int MODE>
__global__ __launch_bounds__(256) void gemm_wmma_kernel(
    const _Float16* __restrict__ A, const _Float16* __restrict__ W,
    const float* __restrict__ bias, _Float16* __restrict__ outh,
    float* __restrict__ outf) {
  // halves: A buffers [0,2048) [2048,4096); B buffers [4096,12288) [12288,20480)
  __shared__ __align__(64) _Float16 smem[2 * 64 * 32 + 2 * 256 * 32];
  const unsigned smembase = (unsigned)(size_t)(const void*)&smem[0];

  const int t = threadIdx.x;
  const int lane = t & 31;
  const int wave = t >> 5;
  const int bid = blockIdx.x;              // 256 blocks
  const int mblk = (bid >> 2) * 64;        // 64 M-blocks
  const int nblk = (bid & 3) * 256;        // 4  N-blocks
  const int wm = wave >> 2;                // 0..1
  const int wn = wave & 3;                 // 0..3
  const int lh = lane & 15;
  const int hi = lane >> 4;
  const int akoff = hi ? 8 : 0;
  const int bkoff = hi ? 16 : 0;

  const _Float16* Ag = A + (size_t)mblk * DIMC;
  const _Float16* Wg = W + (size_t)nblk * DIMC;

  // per-thread DMA slice: one 16B chunk of A, four 16B chunks of B
  const int crow = t >> 2;
  const int cq = (t & 3) * 8;

  auto stage = [&](int kc, int buf) {
    const _Float16* Ak = Ag + kc * 32;
    const _Float16* Wk = Wg + kc * 32;
    async_b128(smembase + (unsigned)(buf * 2048 + crow * 32 + cq) * 2,
               Ak + (size_t)crow * DIMC + cq);
#pragma unroll
    for (int i = 0; i < 4; ++i) {
      const int brow = crow + i * 64;
      async_b128(smembase + (unsigned)(4096 + buf * 8192 + brow * 32 + cq) * 2,
                 Wk + (size_t)brow * DIMC + cq);
    }
  };

  v8f acc[2][4];
  v8f z = {};
#pragma unroll
  for (int i = 0; i < 2; ++i)
#pragma unroll
    for (int j = 0; j < 4; ++j) acc[i][j] = z;

  stage(0, 0);  // prologue: fill buffer 0 with k-chunk 0

  for (int kt = 0; kt < DIMC / 32; ++kt) {
    const int cur = kt & 1;
    wait_async0();        // this wave's DMA into `cur` is complete
    __syncthreads();      // all waves done waiting AND done reading `1-cur`
    if (kt + 1 < DIMC / 32) stage(kt + 1, 1 - cur);  // overlap next chunk

    const unsigned abase = cur * 2048;
    const unsigned bbase = 4096 + cur * 8192;
    v16h afr[2], bfr[4];
#pragma unroll
    for (int i = 0; i < 2; ++i)
      afr[i] = load_afrag(&smem[abase + (wm * 32 + i * 16 + lh) * 32], akoff);
#pragma unroll
    for (int j = 0; j < 4; ++j)
      bfr[j] = *(const v16h*)&smem[bbase + (wn * 64 + j * 16 + lh) * 32 + bkoff];
#pragma unroll
    for (int i = 0; i < 2; ++i)
#pragma unroll
      for (int j = 0; j < 4; ++j)
        acc[i][j] = wmma_f16(afr[i], bfr[j], acc[i][j]);
  }

#pragma unroll
  for (int i = 0; i < 2; ++i)
#pragma unroll
    for (int j = 0; j < 4; ++j)
#pragma unroll
      for (int r = 0; r < 8; ++r) {
        const int mm = mblk + wm * 32 + i * 16 + r + (hi ? 8 : 0);
        const int nn = nblk + wn * 64 + j * 16 + lh;
        const float val = acc[i][j][r] + bias[nn];
        if (MODE == 2) {
          outf[(size_t)mm * DIMC + nn] = val;
        } else {
          const int b_ = mm >> 11, tt = mm & (TSEQ - 1);
          const int h_ = nn >> 6, d = nn & (HDIM - 1);
          if (MODE == 0) {  // [bh][t][d]
            outh[((size_t)(b_ * NHEAD + h_) * TSEQ + tt) * HDIM + d] =
                (_Float16)val;
          } else {          // [bh][d][t]
            outh[((size_t)(b_ * NHEAD + h_) * HDIM + d) * TSEQ + tt] =
                (_Float16)val;
          }
        }
      }
}

// ---------------------------------------------------------------------------
// Flash attention, causal. One wave handles one (bh, 16-query) tile.
// Per iteration: 32 keys -> 4 WMMA for S (Dh=64 contraction) + 4 WMMA for PV.
// ---------------------------------------------------------------------------
__global__ __launch_bounds__(128) void attn_wmma_kernel(
    const _Float16* __restrict__ q16,  // [bh][t][d]
    const _Float16* __restrict__ k16,  // [bh][t][d]
    const _Float16* __restrict__ vT,   // [bh][d][t]
    _Float16* __restrict__ attn) {     // [b*t][dim] row-major f16
  __shared__ __align__(32) _Float16 plds_all[4 * 16 * 32];
  const int lane = threadIdx.x & 31;
  const int wave = threadIdx.x >> 5;
  _Float16* plds = plds_all + wave * (16 * 32);

  const int gw = blockIdx.x * 4 + wave;  // 4096 waves
  const int bh = gw >> 7;                // batch*head
  const int qt = gw & 127;               // query tile
  const int lh = lane & 15;
  const int hi = lane >> 4;
  const int hi8 = hi ? 8 : 0;
  const int akoff = hi ? 8 : 0;
  const int bkoff = hi ? 16 : 0;

  // Q fragments for this tile: rows qt*16+lh, contraction d in {0..31},{32..63}
  const _Float16* qrow = q16 + ((size_t)bh * TSEQ + qt * 16 + lh) * HDIM;
  const v16h qa0 = load_afrag(qrow, akoff);
  const v16h qa1 = load_afrag(qrow + 32, akoff);

  float Mr[8], Lr[8];
  v8f O[4];
#pragma unroll
  for (int r = 0; r < 8; ++r) { Mr[r] = -3.0e38f; Lr[r] = 0.f; }
  v8f z = {};
#pragma unroll
  for (int d = 0; d < 4; ++d) O[d] = z;

  const int tq_max = qt * 16 + 15;
  for (int j0 = 0; j0 <= tq_max; j0 += 32) {
    // ---- scores: Sa = keys j0..j0+15, Sb = keys j0+16..j0+31 ----
    v8f Sa = z, Sb = z;
    {
      const _Float16* kr0 = k16 + ((size_t)bh * TSEQ + j0 + lh) * HDIM;
      v16h b0 = *(const v16h*)(kr0 + bkoff);        // d = 0..31 slice
      v16h b1 = *(const v16h*)(kr0 + 32 + bkoff);   // d = 32..63 slice
      Sa = wmma_f16(qa0, b0, Sa);
      Sa = wmma_f16(qa1, b1, Sa);
      const _Float16* kr1 = k16 + ((size_t)bh * TSEQ + j0 + 16 + lh) * HDIM;
      v16h c0 = *(const v16h*)(kr1 + bkoff);
      v16h c1 = *(const v16h*)(kr1 + 32 + bkoff);
      Sb = wmma_f16(qa0, c0, Sb);
      Sb = wmma_f16(qa1, c1, Sb);
    }
    // scale + causal mask
    const bool need_mask = (j0 + 31) > (qt * 16);
#pragma unroll
    for (int r = 0; r < 8; ++r) {
      Sa[r] *= 0.125f;  // 1/sqrt(64)
      Sb[r] *= 0.125f;
      if (need_mask) {
        const int tq = qt * 16 + r + hi8;
        if (j0 + lh > tq) Sa[r] = -1.0e30f;
        if (j0 + 16 + lh > tq) Sb[r] = -1.0e30f;
      }
    }
    // ---- online softmax (row = one 16-lane half for fixed r) ----
    float alpha[8];
#pragma unroll
    for (int r = 0; r < 8; ++r) {
      float mx = fmaxf(Sa[r], Sb[r]);
#pragma unroll
      for (int off = 1; off < 16; off <<= 1)
        mx = fmaxf(mx, __shfl_xor(mx, off, 32));
      const float Mn = fmaxf(Mr[r], mx);
      alpha[r] = __expf(Mr[r] - Mn);
      Mr[r] = Mn;
      const float ea = __expf(Sa[r] - Mn);
      const float eb = __expf(Sb[r] - Mn);
      float rsum = ea + eb;
#pragma unroll
      for (int off = 1; off < 16; off <<= 1)
        rsum += __shfl_xor(rsum, off, 32);
      Lr[r] = Lr[r] * alpha[r] + rsum;
      // P tile (row-major q x key) into per-wave LDS for A-layout reload
      plds[(r + hi8) * 32 + lh] = (_Float16)ea;
      plds[(r + hi8) * 32 + 16 + lh] = (_Float16)eb;
      // rescale running output
      O[0][r] *= alpha[r];
      O[1][r] *= alpha[r];
      O[2][r] *= alpha[r];
      O[3][r] *= alpha[r];
    }
    // same-wave DS ordering: drain LDS stores before transposed reload
    asm volatile("s_wait_dscnt 0" ::: "memory");
    const v16h pa = load_afrag(plds + lh * 32, akoff);  // 16x32 A-frag of P
    // ---- O += P @ V  (V^T rows contiguous in key) ----
#pragma unroll
    for (int dblk = 0; dblk < 4; ++dblk) {
      const _Float16* vp =
          vT + ((size_t)bh * HDIM + dblk * 16 + lh) * TSEQ + j0 + bkoff;
      v16h bv = *(const v16h*)vp;
      O[dblk] = wmma_f16(pa, bv, O[dblk]);
    }
  }

  // ---- normalize + store to row-major [b*t][dim] f16 ----
  const int b_ = bh >> 4;
  const int h_ = bh & 15;
#pragma unroll
  for (int r = 0; r < 8; ++r) {
    const float inv = 1.0f / Lr[r];
    const int t = qt * 16 + r + hi8;
#pragma unroll
    for (int dblk = 0; dblk < 4; ++dblk) {
      const int n = h_ * HDIM + dblk * 16 + lh;
      attn[((size_t)b_ * TSEQ + t) * DIMC + n] = (_Float16)(O[dblk][r] * inv);
    }
  }
}

// ---------------------------------------------------------------------------
// Launcher
// ---------------------------------------------------------------------------
extern "C" void kernel_launch(void* const* d_in, const int* in_sizes, int n_in,
                              void* d_out, int out_size, void* d_ws,
                              size_t ws_size, hipStream_t stream) {
  const float* x    = (const float*)d_in[0];
  const float* ln_g = (const float*)d_in[1];
  const float* ln_b = (const float*)d_in[2];
  const float* Wq   = (const float*)d_in[3];
  const float* bq   = (const float*)d_in[4];
  const float* Wk   = (const float*)d_in[5];
  const float* bk   = (const float*)d_in[6];
  const float* Wv   = (const float*)d_in[7];
  const float* bv   = (const float*)d_in[8];
  const float* Wo   = (const float*)d_in[9];
  const float* bo   = (const float*)d_in[10];

  char* ws = (char*)d_ws;
  const size_t MB = 1u << 20;
  _Float16* h16  = (_Float16*)(ws + 0 * MB);   // [4096][1024]    8 MB
  _Float16* wq16 = (_Float16*)(ws + 8 * MB);   // 4x [1024][1024] 8 MB
  _Float16* wk16 = wq16 + 1024 * 1024;
  _Float16* wv16 = wk16 + 1024 * 1024;
  _Float16* wo16 = wv16 + 1024 * 1024;
  _Float16* q16  = (_Float16*)(ws + 16 * MB);  // [bh][t][d]      8 MB
  _Float16* k16  = (_Float16*)(ws + 24 * MB);  // [bh][t][d]      8 MB
  _Float16* vT16 = (_Float16*)(ws + 32 * MB);  // [bh][d][t]      8 MB
  _Float16* a16  = (_Float16*)(ws + 40 * MB);  // [b*t][dim]      8 MB

  cvt4_kernel<<<4096, 256, 0, stream>>>(Wq, Wk, Wv, Wo, wq16, wk16, wv16, wo16);
  ln_f16_kernel<<<BT, 128, 0, stream>>>(x, ln_g, ln_b, h16);

  // 64x256 block tiles: 64 M-blocks x 4 N-blocks = 256 blocks, 256 thr each
  gemm_wmma_kernel<0><<<256, 256, 0, stream>>>(h16, wq16, bq, q16, nullptr);
  gemm_wmma_kernel<0><<<256, 256, 0, stream>>>(h16, wk16, bk, k16, nullptr);
  gemm_wmma_kernel<1><<<256, 256, 0, stream>>>(h16, wv16, bv, vT16, nullptr);

  attn_wmma_kernel<<<1024, 128, 0, stream>>>(q16, k16, vT16, a16);

  gemm_wmma_kernel<2><<<256, 256, 0, stream>>>(a16, wo16, bo, nullptr,
                                               (float*)d_out);
}
